// MoEDet_24137716204032
// MI455X (gfx1250) — matrix operand order repaired and checked by reading, not compiled, not run on hardware
//
#include <hip/hip_runtime.h>

// ---------------- problem constants ----------------
#define E_EXPERTS 23
#define TOPK 3
constexpr int Nt    = 4096;            // B*S tokens
constexpr int Dd    = 1024;            // model dim
constexpr int Hh    = 768;             // hidden dim
constexpr int M_CAP = Nt * TOPK + 128; // padded assignment capacity
constexpr int BM    = 128;             // block tile M
constexpr int BN    = 256;             // block tile N
constexpr int BK    = 32;              // block tile K

// ---------------- types ----------------
typedef __bf16 bf16;
typedef __attribute__((ext_vector_type(16))) __bf16 v16bf;
typedef __attribute__((ext_vector_type(8)))  __bf16 v8bf;
typedef __attribute__((ext_vector_type(8)))  float  v8f;
typedef unsigned int u32x4 __attribute__((ext_vector_type(4)));
typedef int          i32x4 __attribute__((ext_vector_type(4)));
typedef int          i32x8 __attribute__((ext_vector_type(8)));

// ---------------- TDM (Tensor Data Mover) ----------------
#if defined(__has_builtin)
#if __has_builtin(__builtin_amdgcn_tensor_load_to_lds) && \
    __has_builtin(__builtin_amdgcn_s_wait_tensorcnt)
#define HAVE_TDM 1
#endif
#endif

#if defined(HAVE_TDM)
// 2D tile load: tile_d0 contiguous elements (2B each) x tile_d1 rows, row
// stride = stride0 elements; destination LDS is contiguous [tile_d1][tile_d0].
__device__ __forceinline__ void tdm_load_2d(unsigned lds_off,
                                            unsigned long long gaddr,
                                            unsigned tile_d0, unsigned tile_d1,
                                            unsigned long long stride0,
                                            unsigned tensor_d0, unsigned tensor_d1) {
  u32x4 g0;
  g0.x = 1u;                                             // count=1, user mode
  g0.y = lds_off;                                        // LDS byte address
  g0.z = (unsigned)(gaddr & 0xFFFFFFFFull);              // global addr lo
  g0.w = (unsigned)((gaddr >> 32) & 0x01FFFFFFull) | (2u << 30); // hi + type=2
  i32x8 g1;
  const unsigned ds = 1u;                                // data_size: 2 bytes
  g1[0] = (int)(ds << 16);                               // mask=0, flags=0
  g1[1] = (int)((tensor_d0 & 0xFFFFu) << 16);            // dim0[15:0] @ bit48
  g1[2] = (int)((tensor_d0 >> 16) | ((tensor_d1 & 0xFFFFu) << 16));
  g1[3] = (int)((tensor_d1 >> 16) | (tile_d0 << 16));    // tile_dim0 @ bit112
  g1[4] = (int)(tile_d1 & 0xFFFFu);                      // tile_dim1, tile_dim2=0
  g1[5] = (int)(stride0 & 0xFFFFFFFFull);                // dim0 stride @ bit160
  g1[6] = (int)((stride0 >> 32) & 0xFFFFull);            // stride hi, stride1=0
  g1[7] = 0;
  i32x4 gz = {0, 0, 0, 0};
#if __clang_major__ >= 23
  i32x8 gz8 = {0, 0, 0, 0, 0, 0, 0, 0};
  __builtin_amdgcn_tensor_load_to_lds(g0, g1, gz, gz, gz8, 0);
#else
  __builtin_amdgcn_tensor_load_to_lds(g0, g1, gz, gz, 0);
#endif
}
#endif

__device__ __forceinline__ float gelu_exact(float v) {
  return 0.5f * v * (1.0f + erff(v * 0.70710678118654752f));
}

// ---------------- small kernels ----------------
__global__ void zero_small(int* counts, int* cursor, float* Psum) {
  int t = threadIdx.x;
  if (t < E_EXPERTS) { counts[t] = 0; cursor[t] = 0; Psum[t] = 0.f; }
}

__global__ void conv_bf16(const float* __restrict__ in, bf16* __restrict__ out,
                          long long n) {
  long long i = (long long)blockIdx.x * blockDim.x + threadIdx.x;
  if (i < n) out[i] = (bf16)in[i];
}

// in: [E][R][C]  ->  out: [E][C][R]   (bf16 convert + transpose)
__global__ void conv_transpose_bf16(const float* __restrict__ in,
                                    bf16* __restrict__ out,
                                    int Eo, int R, int C) {
  long long total = (long long)Eo * R * C;
  long long i = (long long)blockIdx.x * blockDim.x + threadIdx.x;
  if (i >= total) return;
  int r = (int)(i % R);
  long long t = i / R;
  int c = (int)(t % C);
  int e = (int)(t / C);
  out[((long long)e * C + c) * R + r] = (bf16)in[((long long)e * R + r) * C + c];
}

// one wave per token: gate logits, sigmoid, top-3, combine weights, aux stats
__global__ __launch_bounds__(128)
void gate_topk(const float* __restrict__ x, const float* __restrict__ gW,
               const float* __restrict__ gb, const float* __restrict__ rb,
               int* __restrict__ topk_i, float* __restrict__ topk_w,
               int* __restrict__ counts, float* __restrict__ Psum) {
  const int lane = threadIdx.x & 31;
  const int wave = threadIdx.x >> 5;
  const int n = blockIdx.x * 4 + wave;
  float acc[E_EXPERTS];
#pragma unroll
  for (int e = 0; e < E_EXPERTS; ++e) acc[e] = 0.f;
  const float* xr = x + (size_t)n * Dd;
  for (int d = lane; d < Dd; d += 32) {
    float xv = xr[d];
    const float* wr = gW + (size_t)d * E_EXPERTS;
#pragma unroll
    for (int e = 0; e < E_EXPERTS; ++e) acc[e] += xv * wr[e];
  }
#pragma unroll
  for (int e = 0; e < E_EXPERTS; ++e) {
    float v = acc[e];
    for (int o = 16; o > 0; o >>= 1) v += __shfl_xor(v, o, 32);
    acc[e] = v;
  }
  if (lane == 0) {
    float g[E_EXPERTS]; float gsum = 0.f;
#pragma unroll
    for (int e = 0; e < E_EXPERTS; ++e) {
      g[e] = 1.f / (1.f + __expf(-(acc[e] + gb[e])));
      gsum += g[e];
    }
    int i0 = 0, i1 = 0, i2 = 0;
    float s0 = -1e30f, s1 = -1e30f, s2 = -1e30f;
#pragma unroll
    for (int e = 0; e < E_EXPERTS; ++e) {
      float sc = g[e] + rb[e];
      if (sc > s0)      { s2 = s1; i2 = i1; s1 = s0; i1 = i0; s0 = sc; i0 = e; }
      else if (sc > s1) { s2 = s1; i2 = i1; s1 = sc; i1 = e; }
      else if (sc > s2) { s2 = sc; i2 = e; }
    }
    float w0 = g[i0], w1 = g[i1], w2 = g[i2];
    float inv = 1.f / (w0 + w1 + w2);
    topk_i[n * 3 + 0] = i0; topk_i[n * 3 + 1] = i1; topk_i[n * 3 + 2] = i2;
    topk_w[n * 3 + 0] = w0 * inv; topk_w[n * 3 + 1] = w1 * inv; topk_w[n * 3 + 2] = w2 * inv;
    atomicAdd(&counts[i0], 1); atomicAdd(&counts[i1], 1); atomicAdd(&counts[i2], 1);
    float ginv = 1.f / gsum;
#pragma unroll
    for (int e = 0; e < E_EXPERTS; ++e) atomicAdd(&Psum[e], g[e] * ginv);
  }
}

__global__ void scan_aux(const int* counts, const float* Psum, int* offs,
                         float* aux_out) {
  if (threadIdx.x == 0) {
    int off = 0; float aux = 0.f;
    for (int e = 0; e < E_EXPERTS; ++e) {
      offs[e] = off; off += counts[e];
      float P = Psum[e] / (float)Nt;
      float F = ((float)E_EXPERTS * (float)counts[e]) / (float)(TOPK * Nt);
      aux += P * F;
    }
    offs[E_EXPERTS] = off;
    *aux_out = aux;
  }
}

__global__ __launch_bounds__(128)
void scatter(const int* __restrict__ topk_i, const float* __restrict__ topk_w,
             const int* __restrict__ offs, int* __restrict__ cursor,
             int* __restrict__ tok_ids, float* __restrict__ slot_w) {
  int n = blockIdx.x * 128 + threadIdx.x;
  if (n >= Nt) return;
  for (int k = 0; k < TOPK; ++k) {
    int e = topk_i[n * 3 + k];
    int pos = offs[e] + atomicAdd(&cursor[e], 1);
    tok_ids[pos] = n;
    slot_w[pos] = topk_w[n * 3 + k];
  }
}

// ---------------- WMMA GEMM ----------------
// Block tile 128(M) x 256(N) x 32(K); 8 waves as 2x4; each wave 64x64 =
// 4x4 fragments -> 16 v_wmma_f32_16x16x32_bf16 per K-step per wave.
// Double-buffered LDS; TDM / global loads of tile k+1 overlap compute of k.
// MODE 0: shared up  : A = xb direct (TDM),   epi gelu -> bf16 (h_shared)
// MODE 1: shared down: A = h_shared (TDM),    epi +bias -> f32 store (d_out)
// MODE 2: expert up  : A = xb gathered rows,  epi gelu -> bf16 (h_exp)
// MODE 3: expert down: A = h_exp seg (TDM),   epi (acc+bias)*w -> atomic d_out
template <int MODE>
__global__ __launch_bounds__(256)
void moe_gemm(const bf16* __restrict__ Abase, const bf16* __restrict__ Bbase,
              const float* __restrict__ biasBase, float* __restrict__ outF,
              bf16* __restrict__ outH, const int* __restrict__ token_ids,
              const float* __restrict__ slot_w, const int* __restrict__ e_cnt,
              const int* __restrict__ e_off, int Kdim, int Ncols) {
  __shared__ __align__(64) bf16 sA[2][BM * BK];
  __shared__ __align__(64) bf16 sB[2][BN * BK];

  const int tid = threadIdx.x;
  const int lane = tid & 31;
  const int wave = tid >> 5;
  const int waveM = wave >> 2;   // 0..1 -> 64 rows each
  const int waveN = wave & 3;    // 0..3 -> 64 cols each

  int seg = 0, cnt = 0;
  const bf16* Bp = Bbase;
  const float* bias = biasBase;
  if (MODE >= 2) {
    const int e = blockIdx.z;
    seg = e_off[e];
    cnt = e_cnt[e];
    if ((int)blockIdx.x * BM >= cnt) return;   // uniform early-exit
    Bp = Bbase + (size_t)e * (size_t)Ncols * (size_t)Kdim;
    bias = biasBase + (size_t)e * (size_t)Ncols;
  }

  const int m0 = blockIdx.x * BM;
  const int n0 = blockIdx.y * BN;

  // A-row base for the uniform (TDM-able) modes
  const size_t arow0 = (MODE == 3) ? (size_t)(seg + m0) : (size_t)m0;

  // per-thread copy coordinates: chunk = tid + i*256 -> (r,c) with r=chunk>>2
  const int r0 = tid >> 2;
  const int c0 = (tid & 3) * 8;

  v8f acc[4][4];
#pragma unroll
  for (int i = 0; i < 4; ++i)
#pragma unroll
    for (int j = 0; j < 4; ++j) {
      v8f z = {0.f, 0.f, 0.f, 0.f, 0.f, 0.f, 0.f, 0.f};
      acc[i][j] = z;
    }

#if defined(HAVE_TDM)
  constexpr bool TDM_A = (MODE != 2);
#else
  constexpr bool TDM_A = false;
#endif

  u32x4 pa0 = {0u, 0u, 0u, 0u}, pa1 = {0u, 0u, 0u, 0u};
#if !defined(HAVE_TDM)
  u32x4 pb[4];
#pragma unroll
  for (int i = 0; i < 4; ++i) pb[i] = (u32x4){0u, 0u, 0u, 0u};
#endif

  // ---------- manual A fetch (indexed gather, MODE 2 / fallback) ----------
  auto fetchA = [&](int kt) {
    if (MODE == 2) {
      int ma = m0 + r0, mb = m0 + r0 + 64;
      pa0 = (u32x4){0u, 0u, 0u, 0u};
      pa1 = (u32x4){0u, 0u, 0u, 0u};
      if (ma < cnt) {
        int tk = token_ids[seg + ma];
        pa0 = *(const u32x4*)(Abase + (size_t)tk * Kdim + kt + c0);
      }
      if (mb < cnt) {
        int tk = token_ids[seg + mb];
        pa1 = *(const u32x4*)(Abase + (size_t)tk * Kdim + kt + c0);
      }
    } else {
      pa0 = *(const u32x4*)(Abase + (arow0 + r0) * Kdim + kt + c0);
      pa1 = *(const u32x4*)(Abase + (arow0 + r0 + 64) * Kdim + kt + c0);
    }
  };
  auto commitA = [&](int buf) {
    *(u32x4*)(sA[buf] + r0 * BK + c0) = pa0;
    *(u32x4*)(sA[buf] + (r0 + 64) * BK + c0) = pa1;
  };

  // ---------- tile issue: start loads of tile at kt into buffer buf ----------
  auto issueTile = [&](int kt, int buf) {
#if defined(HAVE_TDM)
    if (tid < 32) {
      if (TDM_A) {
        unsigned long long ga = (unsigned long long)(size_t)(const void*)Abase +
                                2ull * (arow0 * (unsigned)Kdim + (unsigned)kt);
        tdm_load_2d((unsigned)(size_t)(void*)sA[buf], ga, (unsigned)BK,
                    (unsigned)BM, (unsigned long long)(unsigned)Kdim,
                    (unsigned)Kdim, (unsigned)(1u << 30));
      }
      unsigned long long gb = (unsigned long long)(size_t)(const void*)Bp +
                              2ull * ((unsigned long long)(size_t)n0 *
                                      (unsigned)Kdim + (unsigned)kt);
      tdm_load_2d((unsigned)(size_t)(void*)sB[buf], gb, (unsigned)BK,
                  (unsigned)BN, (unsigned long long)(unsigned)Kdim,
                  (unsigned)Kdim, (unsigned)Ncols);
    }
    if (!TDM_A) fetchA(kt);
#else
    fetchA(kt);
#pragma unroll
    for (int i = 0; i < 4; ++i)
      pb[i] = *(const u32x4*)(Bp + (size_t)(n0 + r0 + i * 64) * Kdim + kt + c0);
#endif
  };
  // ---------- tile finish: drain everything into buffer buf ----------
  auto finishTile = [&](int buf) {
    if (!TDM_A) commitA(buf);
#if defined(HAVE_TDM)
    if (tid < 32) __builtin_amdgcn_s_wait_tensorcnt(0);
#else
#pragma unroll
    for (int i = 0; i < 4; ++i)
      *(u32x4*)(sB[buf] + (r0 + i * 64) * BK + c0) = pb[i];
#endif
  };

  // ---------- prologue ----------
  issueTile(0, 0);
  finishTile(0);
  __syncthreads();

  int p = 0;
  for (int kt = 0; kt < Kdim; kt += BK) {
    const bool hasNext = (kt + BK) < Kdim;
    const int q = p ^ 1;
    if (hasNext) issueTile(kt + BK, q);   // DMA/loads overlap compute below

    // ---- compute from buffer p (A and B fragments: contiguous K per lane half) ----
    {
      const int ml = lane & 15;
      const int hi = lane >> 4;       // lane half
      const int base1 = hi * 16;      // A: K 0..15 vs 16..31, contiguous
      union FragA { v16bf v; v8bf h[2]; } a[4];
#pragma unroll
      for (int mi = 0; mi < 4; ++mi) {
        int row = waveM * 64 + mi * 16 + ml;
        a[mi].h[0] = *(const v8bf*)(sA[p] + row * BK + base1);
        a[mi].h[1] = *(const v8bf*)(sA[p] + row * BK + base1 + 8);
      }
      v16bf b[4];
      const int kb = hi * 16;         // B: K 0..15 vs 16..31, contiguous
#pragma unroll
      for (int ni = 0; ni < 4; ++ni) {
        int coln = waveN * 64 + ni * 16 + ml;
        b[ni] = *(const v16bf*)(sB[p] + coln * BK + kb);
      }
#pragma unroll
      for (int mi = 0; mi < 4; ++mi)
#pragma unroll
        for (int ni = 0; ni < 4; ++ni)
          acc[mi][ni] = __builtin_amdgcn_wmma_f32_16x16x32_bf16(
              false, a[mi].v, false, b[ni], (short)0, acc[mi][ni], false, false);
    }

    if (hasNext) finishTile(q);
    __syncthreads();
    p = q;
  }

  // ---- epilogue: VGPR r -> M = r + (lane<16 ? 0 : 8) ----
  const int ml = lane & 15;
  const int mhalf = (lane >> 4) * 8;
#pragma unroll
  for (int mi = 0; mi < 4; ++mi) {
#pragma unroll
    for (int ni = 0; ni < 4; ++ni) {
      int gn = n0 + waveN * 64 + ni * 16 + ml;
      float bv = bias[gn];
#pragma unroll
      for (int r = 0; r < 8; ++r) {
        int m = m0 + waveM * 64 + mi * 16 + mhalf + r;  // tile-space row
        float v = acc[mi][ni][r] + bv;
        if (MODE == 0) {
          outH[(size_t)m * Ncols + gn] = (bf16)gelu_exact(v);
        } else if (MODE == 1) {
          outF[(size_t)m * Ncols + gn] = v;
        } else if (MODE == 2) {
          if (m < cnt) outH[(size_t)(seg + m) * Ncols + gn] = (bf16)gelu_exact(v);
        } else {  // MODE == 3
          if (m < cnt) {
            int tk = token_ids[seg + m];
            float w = slot_w[seg + m];
            atomicAdd(&outF[(size_t)tk * Ncols + gn], v * w);
          }
        }
      }
    }
  }
}

// ---------------- host launcher ----------------
extern "C" void kernel_launch(void* const* d_in, const int* in_sizes, int n_in,
                              void* d_out, int out_size, void* d_ws,
                              size_t ws_size, hipStream_t stream) {
  const float* x   = (const float*)d_in[0];
  const float* gW  = (const float*)d_in[1];
  const float* gb  = (const float*)d_in[2];
  const float* rb  = (const float*)d_in[3];
  const float* W1  = (const float*)d_in[4];
  const float* b1  = (const float*)d_in[5];
  const float* W2  = (const float*)d_in[6];
  const float* b2  = (const float*)d_in[7];
  const float* sW1 = (const float*)d_in[8];
  const float* sb1 = (const float*)d_in[9];
  const float* sW2 = (const float*)d_in[10];
  const float* sb2 = (const float*)d_in[11];
  float* out = (float*)d_out;

  char* ws = (char*)d_ws;
  size_t off = 0;
  auto alloc = [&](size_t bytes) -> char* {
    char* p = ws + off;
    off = (off + bytes + 255) & ~(size_t)255;
    return p;
  };
  bf16*  xb     = (bf16*)alloc((size_t)Nt * Dd * 2);
  bf16*  w1t    = (bf16*)alloc((size_t)E_EXPERTS * Hh * Dd * 2);  // [E][H][D]
  bf16*  w2t    = (bf16*)alloc((size_t)E_EXPERTS * Dd * Hh * 2);  // [E][D][H]
  bf16*  sw1t   = (bf16*)alloc((size_t)Hh * Dd * 2);              // [H][D]
  bf16*  sw2t   = (bf16*)alloc((size_t)Dd * Hh * 2);              // [D][H]
  bf16*  hsh    = (bf16*)alloc((size_t)Nt * Hh * 2);
  bf16*  hex    = (bf16*)alloc((size_t)M_CAP * Hh * 2);
  int*   tok    = (int*)alloc((size_t)M_CAP * 4);
  float* swgt   = (float*)alloc((size_t)M_CAP * 4);
  int*   tki    = (int*)alloc((size_t)Nt * 3 * 4);
  float* tkw    = (float*)alloc((size_t)Nt * 3 * 4);
  int*   counts = (int*)alloc(E_EXPERTS * 4);
  int*   cursor = (int*)alloc(E_EXPERTS * 4);
  int*   offs   = (int*)alloc((E_EXPERTS + 1) * 4);
  float* Ps     = (float*)alloc(E_EXPERTS * 4);

  zero_small<<<1, 64, 0, stream>>>(counts, cursor, Ps);

  {
    long long n = (long long)Nt * Dd;
    conv_bf16<<<(unsigned)((n + 255) / 256), 256, 0, stream>>>(x, xb, n);
  }
  {
    long long n = (long long)E_EXPERTS * Dd * Hh;
    unsigned g = (unsigned)((n + 255) / 256);
    conv_transpose_bf16<<<g, 256, 0, stream>>>(W1, w1t, E_EXPERTS, Dd, Hh);
    conv_transpose_bf16<<<g, 256, 0, stream>>>(W2, w2t, E_EXPERTS, Hh, Dd);
  }
  {
    long long n = (long long)Dd * Hh;
    unsigned g = (unsigned)((n + 255) / 256);
    conv_transpose_bf16<<<g, 256, 0, stream>>>(sW1, sw1t, 1, Dd, Hh);
    conv_transpose_bf16<<<g, 256, 0, stream>>>(sW2, sw2t, 1, Hh, Dd);
  }

  gate_topk<<<Nt / 4, 128, 0, stream>>>(x, gW, gb, rb, tki, tkw, counts, Ps);
  scan_aux<<<1, 32, 0, stream>>>(counts, Ps, offs, out + (size_t)Nt * Dd);
  scatter<<<Nt / 128, 128, 0, stream>>>(tki, tkw, offs, cursor, tok, swgt);

  // shared MLP (also fully (re)writes d_out before expert atomics)
  moe_gemm<0><<<dim3(Nt / BM, Hh / BN, 1), 256, 0, stream>>>(
      xb, sw1t, sb1, nullptr, hsh, nullptr, nullptr, nullptr, nullptr, Dd, Hh);
  moe_gemm<1><<<dim3(Nt / BM, Dd / BN, 1), 256, 0, stream>>>(
      hsh, sw2t, sb2, out, nullptr, nullptr, nullptr, nullptr, nullptr, Hh, Dd);

  // routed experts (sparse, per-expert segments)
  moe_gemm<2><<<dim3(Nt / BM, Hh / BN, E_EXPERTS), 256, 0, stream>>>(
      xb, w1t, b1, nullptr, hex, tok, swgt, counts, offs, Dd, Hh);
  moe_gemm<3><<<dim3(Nt / BM, Dd / BN, E_EXPERTS), 256, 0, stream>>>(
      hex, w2t, b2, out, nullptr, tok, swgt, counts, offs, Hh, Dd);
}
